// DeFTAttention_13993003451042
// MI455X (gfx1250) — compile-verified, matrix operand order checked by reading, not run on hardware
//
#include <hip/hip_runtime.h>
#include <cstddef>

// ---------------------------------------------------------------------------
// DeFT tree attention, flash-decoding style, for MI455X (gfx1250, wave32).
//   q:    f32 [64, 32*128]
//   k,v:  f32 [32768, 8, 128]   (GQA: query head h uses kv head h/4)
//   mask: bool (1 byte/elem, JAX bool_) [64, 32768]
//   out:  f32 [64, 32*128]
// Compute-bound (AI ~930 FLOP/B): cast to f16 on load, use
// v_wmma_f32_16x16x32_f16 for both QK^T and PV.
// ---------------------------------------------------------------------------

#define NHEADS 32
#define NKVH 8
#define HDIM 128
#define NQ 64
#define KVLEN 32768
#define GROUP 4           // NHEADS / NKVH
#define QSTRIDE (NHEADS * HDIM)
#define KV_TILE 32
#define SM_SCALE 0.08838834764831845f   // 1/sqrt(128)

typedef __attribute__((ext_vector_type(16))) _Float16 v16h;
typedef __attribute__((ext_vector_type(8)))  float    v8f;

union Frag {
  v16h v;
  _Float16 h[16];
  unsigned int u[8];      // pair (2j, 2j+1) of halves == u[j]
};

// K-index of the half-pair held in VGPR j for 16-bit A/B operands (ISA 7.12.2):
// lanes 0-15:  j=0..3 -> K=0..7,   j=4..7 -> K=16..23
// lanes 16-31: j=0..3 -> K=8..15,  j=4..7 -> K=24..31
__device__ __forceinline__ int kpair_off(int j, int lhi) {
  return ((j & 4) ? 16 : 0) + (lhi ? 8 : 0) + 2 * (j & 3);
}

__global__ __launch_bounds__(128)
void deft_attn_partial(const float* __restrict__ qg,
                       const float* __restrict__ kg,
                       const float* __restrict__ vg,
                       const unsigned char* __restrict__ maskg,
                       float* __restrict__ O_part,
                       float* __restrict__ M_part,
                       float* __restrict__ L_part,
                       int kv_per_split) {
  // Padded LDS tiles (pads break the 256B-stride bank conflict).
  __shared__ _Float16 k_lds[KV_TILE][HDIM + 8];      // [kv][d]
  __shared__ _Float16 v_lds[HDIM][KV_TILE + 8];      // [d][kv] (transposed)
  __shared__ __align__(16) unsigned char mask_lds[NQ][KV_TILE];
  __shared__ _Float16 p_lds[4][16][KV_TILE + 4];     // per-wave P staging

  const int h     = blockIdx.x;
  const int split = blockIdx.y;
  const int kvh   = h / GROUP;
  const int tid   = threadIdx.x;
  const int w     = tid >> 5;        // wave id == q-tile id (0..3)
  const int lane  = tid & 31;
  const int lhi   = lane >> 4;       // half-wave select
  const int ln    = lane & 15;

  // ---- preload Q fragments (A operand): 4 chunks of K-dim 32 ----
  Frag qf[4];
  {
    const int qrow = w * 16 + ln;    // A-layout: M = lane&15 in both halves
    const float* qp = qg + (size_t)qrow * QSTRIDE + h * HDIM;
    for (int kc = 0; kc < 4; ++kc)
      for (int j = 0; j < 8; ++j) {
        const int d = kc * 32 + kpair_off(j, lhi);
        const float2 f = *(const float2*)(qp + d);
        qf[kc].h[2 * j]     = (_Float16)f.x;
        qf[kc].h[2 * j + 1] = (_Float16)f.y;
      }
  }

  const v8f vzero = {0.f, 0.f, 0.f, 0.f, 0.f, 0.f, 0.f, 0.f};
  v8f acc[8];                          // O tile: 16q x 128d (8 C-frags)
  float m_i[8], l_i[8];
  for (int dt = 0; dt < 8; ++dt) acc[dt] = vzero;
  for (int j = 0; j < 8; ++j) { m_i[j] = -__builtin_inff(); l_i[j] = 0.f; }

  const int kv0    = split * kv_per_split;
  const int ntiles = kv_per_split / KV_TILE;

  for (int t = 0; t < ntiles; ++t) {
    const int kvbase = kv0 + t * KV_TILE;
    __syncthreads();   // previous tile fully consumed before overwrite

    // ---- cooperative global->LDS stage (f32 -> f16 convert) ----
    for (int i = 0; i < 8; ++i) {
      const int idx = tid + i * 128;         // 0..1023 float4s
      const int row = idx >> 5;              // kv row 0..31
      const int d0  = (idx & 31) * 4;
      const size_t gb = ((size_t)(kvbase + row) * NKVH + kvh) * HDIM + d0;
      const float4 kf = *(const float4*)(kg + gb);
      k_lds[row][d0 + 0] = (_Float16)kf.x;
      k_lds[row][d0 + 1] = (_Float16)kf.y;
      k_lds[row][d0 + 2] = (_Float16)kf.z;
      k_lds[row][d0 + 3] = (_Float16)kf.w;
      const float4 vf = *(const float4*)(vg + gb);
      v_lds[d0 + 0][row] = (_Float16)vf.x;
      v_lds[d0 + 1][row] = (_Float16)vf.y;
      v_lds[d0 + 2][row] = (_Float16)vf.z;
      v_lds[d0 + 3][row] = (_Float16)vf.w;
    }
    {   // mask tile: 64 q rows x 32 bytes = 2KB, one int4 per thread
      const int r = tid >> 1;
      const int c = (tid & 1) * 16;
      *(int4*)&mask_lds[r][c] =
          *(const int4*)(maskg + (size_t)r * KVLEN + kvbase + c);
    }
    if (t + 1 < ntiles) {   // global_prefetch_b8 for next tile
      const int row = tid & 31;
      const int dpf = (tid >> 5) * 32;
      const size_t nb = ((size_t)(kvbase + KV_TILE + row) * NKVH + kvh) * HDIM + dpf;
      __builtin_prefetch(kg + nb, 0, 1);
      __builtin_prefetch(vg + nb, 0, 1);
    }
    __syncthreads();

    // ---- scores: S[16q x 32kv] = Q(16x128) x K^T, two 16-col N tiles ----
    v8f s0 = vzero, s1 = vzero;
    for (int kc = 0; kc < 4; ++kc) {
      Frag b0, b1;
      for (int j = 0; j < 8; ++j) {
        const int d = kc * 32 + kpair_off(j, lhi);   // contraction dim
        b0.u[j] = *(const unsigned int*)&k_lds[ln][d];        // N = kv local
        b1.u[j] = *(const unsigned int*)&k_lds[16 + ln][d];
      }
      s0 = __builtin_amdgcn_wmma_f32_16x16x32_f16(false, qf[kc].v, false, b0.v,
                                                  (short)0, s0, false, false);
      s1 = __builtin_amdgcn_wmma_f32_16x16x32_f16(false, qf[kc].v, false, b1.v,
                                                  (short)0, s1, false, false);
    }

    // ---- masked online softmax (rows live across the 16-lane half-wave) ----
    for (int j = 0; j < 8; ++j) {
      const int M  = j + (lhi ? 8 : 0);
      const int qr = w * 16 + M;
      float x0 = s0[j] * SM_SCALE;
      float x1 = s1[j] * SM_SCALE;
      if (!mask_lds[qr][ln])      x0 = -1e9f;
      if (!mask_lds[qr][16 + ln]) x1 = -1e9f;
      float mx = fmaxf(x0, x1);
      mx = fmaxf(mx, __shfl_xor(mx, 8));   // xor<=8 stays in each 16-lane half
      mx = fmaxf(mx, __shfl_xor(mx, 4));
      mx = fmaxf(mx, __shfl_xor(mx, 2));
      mx = fmaxf(mx, __shfl_xor(mx, 1));
      const float mnew  = fmaxf(m_i[j], mx);
      const float alpha = __expf(m_i[j] - mnew);
      const float p0 = __expf(x0 - mnew);
      const float p1 = __expf(x1 - mnew);
      float rs = p0 + p1;
      rs += __shfl_xor(rs, 8);
      rs += __shfl_xor(rs, 4);
      rs += __shfl_xor(rs, 2);
      rs += __shfl_xor(rs, 1);
      l_i[j] = l_i[j] * alpha + rs;
      m_i[j] = mnew;
      for (int dt = 0; dt < 8; ++dt) acc[dt][j] *= alpha;
      p_lds[w][M][ln]      = (_Float16)p0;   // stage P (C-layout -> memory)
      p_lds[w][M][16 + ln] = (_Float16)p1;
    }

    asm volatile("s_wait_dscnt 0" ::: "memory");   // wave-local P visibility

    // ---- O += P(16x32) x V(32x128), 8 N-tiles of 16 along head dim ----
    Frag pf;
    for (int j = 0; j < 8; ++j)
      pf.u[j] = *(const unsigned int*)&p_lds[w][ln][kpair_off(j, lhi)];
    for (int dt = 0; dt < 8; ++dt) {
      Frag vb;
      const int dcol = dt * 16 + ln;                  // N = head-dim column
      for (int j = 0; j < 8; ++j)
        vb.u[j] = *(const unsigned int*)&v_lds[dcol][kpair_off(j, lhi)];
      acc[dt] = __builtin_amdgcn_wmma_f32_16x16x32_f16(false, pf.v, false, vb.v,
                                                       (short)0, acc[dt], false, false);
    }
  }

  // ---- write per-split partials: unnormalized O, running max m, sum l ----
  const size_t base = (size_t)split * NHEADS + h;
  float* Op = O_part + base * NQ * HDIM;
  for (int dt = 0; dt < 8; ++dt)
    for (int j = 0; j < 8; ++j) {
      const int qr = w * 16 + j + (lhi ? 8 : 0);
      const int d  = dt * 16 + ln;
      Op[(size_t)qr * HDIM + d] = acc[dt][j];
    }
  if (ln == 0) {   // lane 0 -> rows 0..7, lane 16 -> rows 8..15
    for (int j = 0; j < 8; ++j) {
      const int qr = w * 16 + j + (lhi ? 8 : 0);
      M_part[base * NQ + qr] = m_i[j];
      L_part[base * NQ + qr] = l_i[j];
    }
  }
}

__global__ __launch_bounds__(128)
void deft_attn_combine(const float* __restrict__ O_part,
                       const float* __restrict__ M_part,
                       const float* __restrict__ L_part,
                       float* __restrict__ out, int S) {
  const int h  = blockIdx.x;
  const int qr = blockIdx.y;
  const int d  = threadIdx.x;
  float mg = -__builtin_inff();
  for (int s = 0; s < S; ++s)
    mg = fmaxf(mg, M_part[((size_t)s * NHEADS + h) * NQ + qr]);
  float num = 0.f, den = 0.f;
  for (int s = 0; s < S; ++s) {
    const size_t b = (size_t)s * NHEADS + h;
    const float wgt = __expf(M_part[b * NQ + qr] - mg);
    den += wgt * L_part[b * NQ + qr];
    num += wgt * O_part[(b * NQ + qr) * HDIM + d];
  }
  out[(size_t)qr * QSTRIDE + h * HDIM + d] = num / den;
}

extern "C" void kernel_launch(void* const* d_in, const int* in_sizes, int n_in,
                              void* d_out, int out_size, void* d_ws, size_t ws_size,
                              hipStream_t stream) {
  const float* q = (const float*)d_in[0];
  const float* k = (const float*)d_in[1];
  const float* v = (const float*)d_in[2];
  const unsigned char* mask = (const unsigned char*)d_in[3];  // JAX bool_: 1B/elem
  float* out = (float*)d_out;

  // Flash-decoding split count: largest power of two <= 32 fitting workspace.
  const size_t per_split =
      ((size_t)NHEADS * NQ * HDIM + 2 * (size_t)NHEADS * NQ) * sizeof(float);
  int S = 32;
  while (S > 1 && (size_t)S * per_split > ws_size) S >>= 1;

  float* O_part = (float*)d_ws;
  float* M_part = O_part + (size_t)S * NHEADS * NQ * HDIM;
  float* L_part = M_part + (size_t)S * NHEADS * NQ;
  const int kv_per_split = KVLEN / S;

  deft_attn_partial<<<dim3(NHEADS, S), dim3(128), 0, stream>>>(
      q, k, v, mask, O_part, M_part, L_part, kv_per_split);
  deft_attn_combine<<<dim3(NHEADS, NQ), dim3(HDIM), 0, stream>>>(
      O_part, M_part, L_part, out, S);
}
